// MultiVariateHawkesProcessModel_62139586838986
// MI455X (gfx1250) — matrix operand order repaired and checked by reading, not compiled, not run on hardware
//
#include <hip/hip_runtime.h>
#include <hip/hip_bf16.h>
#include <math.h>

typedef __attribute__((ext_vector_type(16))) _Float16 v16h;
typedef __attribute__((ext_vector_type(8)))  float    v8f;

#define NUM_TYPE 50
#define BATCH    8
#define NEV      2048
#define VPAD     64          // type dim padded to 4 x 16 WMMA column blocks
#define ROWS_PER_BLOCK 128   // 8 waves x 16-row strips

__device__ __forceinline__ float softplus_f(float x) {
    return (x > 20.f) ? x : log1pf(__expf(x));
}

// ---------------------------------------------------------------------------
// One-hot fill: Et[b][v][j] = 1[et_j == v], f16, column-major (j contiguous).
// Batch-invariant w.r.t. row tiles -> computed once, reused by 16 blocks/batch.
// ---------------------------------------------------------------------------
__global__ void hawkes_onehot_fill_kernel(const int* __restrict__ event_types,
                                          _Float16* __restrict__ Et)
{
    const int row = blockIdx.x;          // row = b*VPAD + v
    const int b   = row >> 6;
    const int v   = row & 63;
    for (int j = threadIdx.x; j < NEV; j += 256) {
        int et = event_types[b * NEV + j] - 1;
        et = min(max(et, 0), NUM_TYPE - 1);
        Et[(size_t)row * NEV + j] = (_Float16)((et == v) ? 1.0f : 0.0f);
    }
}

// ---------------------------------------------------------------------------
// Prep kernel: softplus params, compensator (O(N) per batch), zero loglik acc.
// ---------------------------------------------------------------------------
__global__ void hawkes_prep_kernel(const float* __restrict__ event_times,
                                   const int*   __restrict__ event_types,
                                   const float* __restrict__ input_mask,
                                   const float* __restrict__ t0,
                                   const float* __restrict__ t1,
                                   const float* __restrict__ mu,
                                   const float* __restrict__ alpha,
                                   const float* __restrict__ beta,
                                   float* __restrict__ out)
{
    __shared__ float colsum[NUM_TYPE];
    __shared__ float red[256];
    const int b   = blockIdx.x;
    const int tid = threadIdx.x;
    const float omega = softplus_f(beta[0]);

    if (tid < NUM_TYPE) {
        float s = 0.f;
        for (int u = 0; u < NUM_TYPE; ++u) s += softplus_f(alpha[u * NUM_TYPE + tid]);
        colsum[tid] = s;   // sum_u Ahat[u, tid]
    }
    __syncthreads();

    const float tend = t1[b];
    float local = 0.f;
    for (int i = tid; i < NEV; i += 256) {
        int et = event_types[b * NEV + i] - 1;
        et = min(max(et, 0), NUM_TYPE - 1);
        float ik = 1.f - __expf(-omega * (tend - event_times[b * NEV + i]));
        local += colsum[et] * ik * input_mask[b * NEV + i];
    }
    red[tid] = local;
    __syncthreads();
    for (int s = 128; s > 0; s >>= 1) {
        if (tid < s) red[tid] += red[tid + s];
        __syncthreads();
    }
    if (tid == 0) {
        float msum = 0.f;
        for (int u = 0; u < NUM_TYPE; ++u) msum += softplus_f(mu[u]);
        out[BATCH + b] = (tend - t0[b]) * msum + red[0];  // compensator
        out[b] = 0.f;                                     // loglik accumulator
    }
}

// ---------------------------------------------------------------------------
// A fragment (16x32 f16) via chunk-factored kernel:
//   K[i,j] = [w*exp(w(tref - t_i))] * cf[j],  cf[j] = exp(w(t_j - t[j|31]))
// One v_exp per lane per chunk; 16 LDS loads + 16 muls for the elements.
// A k order used here (B matches): lane = M (+16 lane group = upper K half), e -> K index.
// ---------------------------------------------------------------------------
template<bool MASKED>
__device__ __forceinline__ v16h make_a_frag(const float* __restrict__ t_lds,
                                            const float* __restrict__ cf_lds,
                                            int j0, int grp, int m_row,
                                            float t_ia, float omega)
{
    const float tref   = t_lds[j0 + 31];
    const float rowfac = omega * __expf(omega * (tref - t_ia)); // arg <= 0 steady-state
    v16h a;
    #pragma unroll
    for (int e = 0; e < 16; ++e) {
        int k = e + (grp << 4);
        int j = j0 + k;
        float val = rowfac * cf_lds[j];
        if (MASKED && !(m_row > j)) val = 0.f;
        a[e] = (_Float16)val;
    }
    return a;
}

// ---------------------------------------------------------------------------
// Main kernel: rates via strict-lower-triangular (N x N) x (N x 64) WMMA GEMM.
//   H = K_tril * E  (f16 operands, f32 accumulation), rate_i = mhat + <G_i,H_i>
// PRECOMP path: B fragments from precomputed column-major Et, double-buffered
// one chunk ahead so global_load latency overlaps the 4 WMMAs.
// ---------------------------------------------------------------------------
template<bool PRECOMP>
__global__ void __launch_bounds__(256, 1)
hawkes_rates_wmma_kernel(const float* __restrict__ event_times,
                         const int*   __restrict__ event_types,
                         const float* __restrict__ input_mask,
                         const float* __restrict__ mu,
                         const float* __restrict__ alpha,
                         const float* __restrict__ beta,
                         const _Float16* __restrict__ Et,
                         float* __restrict__ out)
{
    __shared__ float t_lds[NEV];
    __shared__ float cf_lds[NEV];
    __shared__ int   et_lds[NEV];
    __shared__ float Ahat[NUM_TYPE * VPAD];
    __shared__ float mhat[VPAD];

    const int b   = blockIdx.y;
    const int i0  = blockIdx.x * ROWS_PER_BLOCK;
    const int tid = threadIdx.x;
    const float omega = softplus_f(beta[0]);

    // Phase 1: stage times + clamped 0-based types.
    for (int i = tid; i < NEV; i += 256) {
        t_lds[i] = event_times[b * NEV + i];
        int et = event_types[b * NEV + i] - 1;
        et_lds[i] = min(max(et, 0), NUM_TYPE - 1);
    }
    __syncthreads();
    // Phase 2: chunk-relative decay factors + softplus'd parameters.
    for (int i = tid; i < NEV; i += 256)
        cf_lds[i] = __expf(omega * (t_lds[i] - t_lds[i | 31]));   // in (0,1]
    for (int idx = tid; idx < NUM_TYPE * VPAD; idx += 256) {
        int u = idx >> 6, v = idx & 63;
        Ahat[idx] = (v < NUM_TYPE) ? softplus_f(alpha[u * NUM_TYPE + v]) : 0.f;
    }
    if (tid < VPAD) mhat[tid] = (tid < NUM_TYPE) ? softplus_f(mu[tid]) : 0.f;
    __syncthreads();

    const int lane = tid & 31;          // wave32
    const int wave = __builtin_amdgcn_readfirstlane(tid) >> 5;  // SGPR: scalar branches
    const int grp  = lane >> 4;         // half-wave group
    const int n    = lane & 15;
    const int strip     = i0 + wave * 16;   // this wave's 16 rows (SGPR)
    const int strip_end = strip + 16;
    const int full_end  = strip & ~31;  // chunks below this are fully strict-lower

    const int   m_row = strip + n;          // A-matrix row owned by this lane
    const float t_ia  = t_lds[m_row];

    v8f acc0 = {}, acc1 = {}, acc2 = {}, acc3 = {};

    if (PRECOMP) {
        // Column-major one-hot rows for this lane's 4 column blocks (32B aligned).
        const _Float16* e0 = Et + ((size_t)(b * VPAD + n)      * NEV) + (grp << 4);
        const _Float16* e1 = Et + ((size_t)(b * VPAD + n + 16) * NEV) + (grp << 4);
        const _Float16* e2 = Et + ((size_t)(b * VPAD + n + 32) * NEV) + (grp << 4);
        const _Float16* e3 = Et + ((size_t)(b * VPAD + n + 48) * NEV) + (grp << 4);

        // Prologue: load chunk 0's B fragments.
        v16h b0 = *(const v16h*)(e0);
        v16h b1 = *(const v16h*)(e1);
        v16h b2 = *(const v16h*)(e2);
        v16h b3 = *(const v16h*)(e3);

        #pragma unroll 1
        for (int j0 = 0; j0 < strip_end; j0 += 32) {
            // A fragment: scalar (SGPR) branch keeps EXEC full around WMMA.
            v16h a;
            if (j0 >= full_end) a = make_a_frag<true >(t_lds, cf_lds, j0, grp, m_row, t_ia, omega);
            else                a = make_a_frag<false>(t_lds, cf_lds, j0, grp, m_row, t_ia, omega);

            // Prefetch next chunk's B fragments (overlaps the WMMAs below).
            const int jn = j0 + 32;
            v16h nb0, nb1, nb2, nb3;
            if (jn < strip_end) {
                nb0 = *(const v16h*)(e0 + jn);
                nb1 = *(const v16h*)(e1 + jn);
                nb2 = *(const v16h*)(e2 + jn);
                nb3 = *(const v16h*)(e3 + jn);
            }

            acc0 = __builtin_amdgcn_wmma_f32_16x16x32_f16(false, a, false, b0, (short)0, acc0, false, false);
            acc1 = __builtin_amdgcn_wmma_f32_16x16x32_f16(false, a, false, b1, (short)0, acc1, false, false);
            acc2 = __builtin_amdgcn_wmma_f32_16x16x32_f16(false, a, false, b2, (short)0, acc2, false, false);
            acc3 = __builtin_amdgcn_wmma_f32_16x16x32_f16(false, a, false, b3, (short)0, acc3, false, false);

            if (jn < strip_end) { b0 = nb0; b1 = nb1; b2 = nb2; b3 = nb3; }
        }
    } else {
        // Fallback (no workspace): inline one-hot construction.
        #pragma unroll 1
        for (int j0 = 0; j0 < strip_end; j0 += 32) {
            v16h a;
            if (j0 >= full_end) a = make_a_frag<true >(t_lds, cf_lds, j0, grp, m_row, t_ia, omega);
            else                a = make_a_frag<false>(t_lds, cf_lds, j0, grp, m_row, t_ia, omega);

            v16h b0, b1, b2, b3;
            #pragma unroll
            for (int e = 0; e < 16; ++e) {
                int et = et_lds[j0 + e + (grp << 4)];
                _Float16 one = (_Float16)1.0f, z = (_Float16)0.0f;
                b0[e] = (et == n)      ? one : z;
                b1[e] = (et == n + 16) ? one : z;
                b2[e] = (et == n + 32) ? one : z;
                b3[e] = (et == n + 48) ? one : z;
            }
            acc0 = __builtin_amdgcn_wmma_f32_16x16x32_f16(false, a, false, b0, (short)0, acc0, false, false);
            acc1 = __builtin_amdgcn_wmma_f32_16x16x32_f16(false, a, false, b1, (short)0, acc1, false, false);
            acc2 = __builtin_amdgcn_wmma_f32_16x16x32_f16(false, a, false, b2, (short)0, acc2, false, false);
            acc3 = __builtin_amdgcn_wmma_f32_16x16x32_f16(false, a, false, b3, (short)0, acc3, false, false);
        }
    }

    // Epilogue: C layout -> row = r + 8*grp, col = n; dot with Ahat[et_i, :],
    // butterfly-reduce over the 16-lane group, log, atomic accumulate.
    float lsum = 0.f;
    #pragma unroll
    for (int r = 0; r < 8; ++r) {
        int irow = strip + r + 8 * grp;
        int et   = et_lds[irow];
        const float* Arow = &Ahat[et * VPAD];
        float p = Arow[n]      * acc0[r]
                + Arow[n + 16] * acc1[r]
                + Arow[n + 32] * acc2[r]
                + Arow[n + 48] * acc3[r];
        p += __shfl_xor(p, 1, 32);
        p += __shfl_xor(p, 2, 32);
        p += __shfl_xor(p, 4, 32);
        p += __shfl_xor(p, 8, 32);
        if (n == 0) {
            float rate = mhat[et] + p;
            lsum += logf(rate + 1e-8f) * input_mask[b * NEV + irow];
        }
    }
    if (n == 0) atomicAdd(&out[b], lsum);
}

// ---------------------------------------------------------------------------
extern "C" void kernel_launch(void* const* d_in, const int* in_sizes, int n_in,
                              void* d_out, int out_size, void* d_ws, size_t ws_size,
                              hipStream_t stream)
{
    (void)in_sizes; (void)n_in; (void)out_size;
    const float* event_times = (const float*)d_in[0];
    const int*   event_types = (const int*)  d_in[1];
    const float* input_mask  = (const float*)d_in[2];
    const float* t0          = (const float*)d_in[3];
    const float* t1          = (const float*)d_in[4];
    const float* mu          = (const float*)d_in[5];
    const float* alpha       = (const float*)d_in[6];
    const float* beta        = (const float*)d_in[7];
    float* out = (float*)d_out;   // [0..7] loglik, [8..15] compensator

    const size_t et_bytes = (size_t)BATCH * VPAD * NEV * sizeof(_Float16); // 2 MB
    const bool precomp = (ws_size >= et_bytes);
    _Float16* Et = (_Float16*)d_ws;

    hipLaunchKernelGGL(hawkes_prep_kernel, dim3(BATCH), dim3(256), 0, stream,
                       event_times, event_types, input_mask, t0, t1, mu, alpha, beta, out);

    if (precomp) {
        hipLaunchKernelGGL(hawkes_onehot_fill_kernel,
                           dim3(BATCH * VPAD), dim3(256), 0, stream,
                           event_types, Et);
        hipLaunchKernelGGL(hawkes_rates_wmma_kernel<true>,
                           dim3(NEV / ROWS_PER_BLOCK, BATCH), dim3(256), 0, stream,
                           event_times, event_types, input_mask, mu, alpha, beta, Et, out);
    } else {
        hipLaunchKernelGGL(hawkes_rates_wmma_kernel<false>,
                           dim3(NEV / ROWS_PER_BLOCK, BATCH), dim3(256), 0, stream,
                           event_times, event_types, input_mask, mu, alpha, beta, Et, out);
    }
}
